// SegAttention_66992899883716
// MI455X (gfx1250) — matrix operand order repaired and checked by reading, not compiled, not run on hardware
//
#include <hip/hip_runtime.h>
#include <hip/hip_bf16.h>

typedef __bf16 bf16_t;
typedef __attribute__((ext_vector_type(16))) __bf16 v16bf;
typedef __attribute__((ext_vector_type(8)))  float  v8f;

#define NB 2
#define TT 2048
#define DD 512
#define HH 8
#define HD 64

// Padded LDS row strides (bf16 elements). 72 -> 36 dwords/row (9n mod 16
// covers all residues), 40 -> 20 dwords/row (5n mod 16 covers all residues):
// conflict-free half-wave fragment loads, rows stay 16B-aligned for b128.
#define QSTR 72
#define KSTR 72
#define VSTR 40
#define PSTR 40
#define GSTR 40

__device__ inline unsigned pack2bf(float a, float b) {
    unsigned short la = __builtin_bit_cast(unsigned short, (bf16_t)a);
    unsigned short lb = __builtin_bit_cast(unsigned short, (bf16_t)b);
    return (unsigned)la | ((unsigned)lb << 16);
}

__device__ inline v8f v8f_zero() {
    v8f z;
#pragma unroll
    for (int i = 0; i < 8; ++i) z[i] = 0.0f;
    return z;
}

__device__ inline v8f wmma_bf16(v16bf a, v16bf b, v8f c) {
    return __builtin_amdgcn_wmma_f32_16x16x32_bf16(
        false, a, false, b, (short)0, c, false, false);
}

// CDNA5 async global->LDS copy (16B per lane), tracked by ASYNCcnt.
__device__ inline void async_copy_b128(unsigned lds_byte_off, const bf16_t* gptr) {
    unsigned long long ga = (unsigned long long)(const void*)gptr;
    asm volatile("global_load_async_to_lds_b128 %0, %1, off"
                 :: "v"(lds_byte_off), "v"(ga) : "memory");
}
__device__ inline void wait_async0() {
    asm volatile("s_wait_asynccnt 0" ::: "memory");
}
__device__ inline unsigned lds_off_of(const void* p) {
    return (unsigned)(size_t)p;   // low 32 bits of generic ptr = LDS offset
}

// A fragment: 16x32 bf16, lane = M row, VGPR v holds K pair per ISA table.
__device__ inline v16bf load_frag_a(const bf16_t* base, int stride) {
    const int lane = threadIdx.x & 31;
    const int m = lane & 15;
    const int kofs = (lane < 16) ? 0 : 8;
    union { v16bf v; unsigned u[8]; } f;
#pragma unroll
    for (int v = 0; v < 8; ++v) {
        int k = (v < 4) ? (kofs + 2 * v) : (16 + kofs + 2 * (v - 4));
        f.u[v] = *(const unsigned*)(base + m * stride + k);
    }
    return f.v;
}

// B fragment: 32x16 bf16, lane = N col, VGPR v holds K pair. LDS layout [n][k].
__device__ inline v16bf load_frag_b(const bf16_t* base, int stride) {
    const int lane = threadIdx.x & 31;
    const int n = lane & 15;
    const int kb = (lane < 16) ? 0 : 8;
    union { v16bf v; unsigned u[8]; } f;
#pragma unroll
    for (int v = 0; v < 8; ++v) {
        int k = (v < 4) ? (kb + 2 * v) : (16 + kb + 2 * (v - 4));
        f.u[v] = *(const unsigned*)(base + n * stride + k);
    }
    return f.v;
}

// ---------------------------------------------------------------------------
// Kernel 0: softmax over head_dim of the learned per-head scale (8x64 values)
__global__ void scale_softmax_kernel(const float* __restrict__ a,
                                     const float* __restrict__ b,
                                     float* __restrict__ oa,
                                     float* __restrict__ ob) {
    const float* src = (blockIdx.y == 0) ? a : b;
    float* dst = (blockIdx.y == 0) ? oa : ob;
    const int h = blockIdx.x;
    const int d = threadIdx.x;
    float mx = -1e30f;
    for (int i = 0; i < HD; ++i) mx = fmaxf(mx, src[h * HD + i]);
    float sum = 0.0f;
    for (int i = 0; i < HD; ++i) sum += __expf(src[h * HD + i] - mx);
    dst[h * HD + d] = __expf(src[h * HD + d] - mx) / sum;
}

// ---------------------------------------------------------------------------
// Kernel 1: QKV projection  C[M=NB*T, 3D] = X @ W^T  (bf16 WMMA, f32 acc)
#define GM 128
#define GN 64
#define GK 32
__global__ __launch_bounds__(256)
void qkv_gemm_kernel(const float* __restrict__ X, const float* __restrict__ W,
                     bf16_t* __restrict__ qout, bf16_t* __restrict__ kout,
                     bf16_t* __restrict__ vout, const float* __restrict__ qscale) {
    __shared__ __align__(16) bf16_t Alds[GM * GSTR];
    __shared__ __align__(16) bf16_t Blds[GN * GSTR];
    const int tid  = threadIdx.x;
    const int wave = tid >> 5;
    const int lane = tid & 31;
    const int blkM = blockIdx.x * GM;
    const int blkN = blockIdx.y * GN;

    v8f acc[4];
#pragma unroll
    for (int t = 0; t < 4; ++t) acc[t] = v8f_zero();

    for (int kb = 0; kb < DD; kb += GK) {
        __syncthreads();
        // A tile: 128x32 f32 -> bf16, 1024 float4, 4 per thread
#pragma unroll
        for (int i = 0; i < 4; ++i) {
            int f = tid + i * 256;
            int row = f >> 3, c4 = f & 7;
            float4 x4 = *(const float4*)(X + (size_t)(blkM + row) * DD + kb + c4 * 4);
            unsigned* dst = (unsigned*)(Alds + row * GSTR + c4 * 4);
            dst[0] = pack2bf(x4.x, x4.y);
            dst[1] = pack2bf(x4.z, x4.w);
        }
        // B tile: 64x32 (rows of W), 512 float4, 2 per thread
#pragma unroll
        for (int i = 0; i < 2; ++i) {
            int f = tid + i * 256;
            int row = f >> 3, c4 = f & 7;
            float4 w4 = *(const float4*)(W + (size_t)(blkN + row) * DD + kb + c4 * 4);
            unsigned* dst = (unsigned*)(Blds + row * GSTR + c4 * 4);
            dst[0] = pack2bf(w4.x, w4.y);
            dst[1] = pack2bf(w4.z, w4.w);
        }
        __syncthreads();
        v16bf a = load_frag_a(Alds + wave * 16 * GSTR, GSTR);
#pragma unroll
        for (int t = 0; t < 4; ++t) {
            v16bf b = load_frag_b(Blds + t * 16 * GSTR, GSTR);
            acc[t] = wmma_bf16(a, b, acc[t]);
        }
    }

    const int rbase = blkM + wave * 16 + ((lane < 16) ? 0 : 8);
#pragma unroll
    for (int t = 0; t < 4; ++t) {
        int cbase = blkN + t * 16 + (lane & 15);
        int which = cbase >> 9;           // 0=q 1=k 2=v
        int h = (cbase & 511) >> 6;
        int d = cbase & 63;
        bf16_t* outp = (which == 0) ? qout : ((which == 1) ? kout : vout);
        float sc = (which == 0) ? qscale[cbase & 511] : 1.0f;
#pragma unroll
        for (int r = 0; r < 8; ++r) {
            int row = rbase + r;
            int n = row >> 11;            // / T
            int trow = row & (TT - 1);
            size_t off = (((size_t)(n * HH + h) * TT + trow) << 6) + d;
            outp[off] = (bf16_t)(acc[t][r] * sc);
        }
    }
}

// ---------------------------------------------------------------------------
// Kernel 2: dual attention, flash-style two-pass, per (n,h) head.
// grid (T/64, NB*HH), block 128 (4 waves, 16 query rows per wave).
#define BM 64
#define BKEY 32
__global__ __launch_bounds__(128)
void attn_kernel(const bf16_t* __restrict__ Q, const bf16_t* __restrict__ K,
                 const bf16_t* __restrict__ Vc, const bf16_t* __restrict__ Vs,
                 float* __restrict__ Out, const float* __restrict__ sp) {
    __shared__ __align__(16) bf16_t Qlds[BM * QSTR];
    __shared__ __align__(16) bf16_t Klds[BKEY * KSTR];
    __shared__ __align__(16) bf16_t Vlds[HD * VSTR];        // transposed [d][key]
    __shared__ __align__(16) bf16_t Plds[4 * 16 * PSTR];    // per-wave P staging

    const int tid  = threadIdx.x;
    const int lane = tid & 31;
    const int wave = tid >> 5;
    const int nh   = blockIdx.y;
    const int rowblk = blockIdx.x * BM;
    const size_t base = (size_t)nh * TT * HD;

    const unsigned q0 = lds_off_of(Qlds);
    const unsigned k0 = lds_off_of(Klds);

    // Q tile 64x64 bf16: 512 async 16B chunks, 4 per thread
#pragma unroll
    for (int i = 0; i < 4; ++i) {
        int f = tid + i * 128;
        int row = f >> 3, c = f & 7;
        async_copy_b128(q0 + (unsigned)(row * QSTR + c * 8) * 2,
                        Q + base + (size_t)(rowblk + row) * HD + c * 8);
    }

    bf16_t* Pw = Plds + wave * 16 * PSTR;
    const bf16_t* Qw = Qlds + wave * 16 * QSTR;

    float M[8], Z[8], U[8];
    v8f Oc[4], Oa[4];
#pragma unroll
    for (int r = 0; r < 8; ++r) { M[r] = -1e30f; Z[r] = 0.0f; U[r] = 0.0f; }
#pragma unroll
    for (int t = 0; t < 4; ++t) { Oc[t] = v8f_zero(); Oa[t] = v8f_zero(); }

    // ------------- pass 1: online softmax + attn @ V_cross -------------
    for (int jc = 0; jc < TT; jc += BKEY) {
        __syncthreads();
        // K tile 32x64 [key][d]: 256 async 16B chunks, 2 per thread
#pragma unroll
        for (int i = 0; i < 2; ++i) {
            int f = tid + i * 128;
            int row = f >> 3, c = f & 7;
            async_copy_b128(k0 + (unsigned)(row * KSTR + c * 8) * 2,
                            K + base + (size_t)(jc + row) * HD + c * 8);
        }
        // V tile transposed to [d][key] (manual: element scatter)
#pragma unroll
        for (int i = 0; i < 2; ++i) {
            int f = tid + i * 128;
            int key = f >> 3, db = (f & 7) * 8;
            uint4 v4 = *(const uint4*)(Vc + base + (size_t)(jc + key) * HD + db);
            const bf16_t* e = (const bf16_t*)&v4;
#pragma unroll
            for (int j = 0; j < 8; ++j) Vlds[(db + j) * VSTR + key] = e[j];
        }
        wait_async0();
        __syncthreads();

        v8f s0 = v8f_zero(), s1 = v8f_zero();
#pragma unroll
        for (int kk = 0; kk < 2; ++kk) {
            v16bf a  = load_frag_a(Qw + kk * 32, QSTR);
            v16bf b0 = load_frag_b(Klds + kk * 32, KSTR);
            v16bf b1 = load_frag_b(Klds + 16 * KSTR + kk * 32, KSTR);
            s0 = wmma_bf16(a, b0, s0);
            s1 = wmma_bf16(a, b1, s1);
        }
#pragma unroll
        for (int r = 0; r < 8; ++r) {
            float mx = fmaxf(s0[r], s1[r]);
#pragma unroll
            for (int m = 1; m < 16; m <<= 1) mx = fmaxf(mx, __shfl_xor(mx, m, 32));
            float Mn = fmaxf(M[r], mx);
            float al = __expf(M[r] - Mn);
            float e0 = __expf(s0[r] - Mn), e1 = __expf(s1[r] - Mn);
            float ps = e0 + e1;
#pragma unroll
            for (int m = 1; m < 16; m <<= 1) ps += __shfl_xor(ps, m, 32);
            Z[r] = Z[r] * al + ps;
            M[r] = Mn;
            int prow = r + ((lane < 16) ? 0 : 8);
            int pcol = lane & 15;
            Pw[prow * PSTR + pcol]      = (bf16_t)e0;
            Pw[prow * PSTR + 16 + pcol] = (bf16_t)e1;
#pragma unroll
            for (int t = 0; t < 4; ++t) Oc[t][r] *= al;
        }
        v16bf pa = load_frag_a(Pw, PSTR);
#pragma unroll
        for (int t = 0; t < 4; ++t) {
            v16bf b = load_frag_b(Vlds + t * 16 * VSTR, VSTR);
            Oc[t] = wmma_bf16(pa, b, Oc[t]);
        }
    }

    float Zinv[8];
#pragma unroll
    for (int r = 0; r < 8; ++r) Zinv[r] = 1.0f / Z[r];

    // ------------- pass 2: anti = softmax(-attn), anti @ V_self -------------
    for (int jc = 0; jc < TT; jc += BKEY) {
        __syncthreads();
#pragma unroll
        for (int i = 0; i < 2; ++i) {
            int f = tid + i * 128;
            int row = f >> 3, c = f & 7;
            async_copy_b128(k0 + (unsigned)(row * KSTR + c * 8) * 2,
                            K + base + (size_t)(jc + row) * HD + c * 8);
        }
#pragma unroll
        for (int i = 0; i < 2; ++i) {
            int f = tid + i * 128;
            int key = f >> 3, db = (f & 7) * 8;
            uint4 v4 = *(const uint4*)(Vs + base + (size_t)(jc + key) * HD + db);
            const bf16_t* e = (const bf16_t*)&v4;
#pragma unroll
            for (int j = 0; j < 8; ++j) Vlds[(db + j) * VSTR + key] = e[j];
        }
        wait_async0();
        __syncthreads();

        v8f s0 = v8f_zero(), s1 = v8f_zero();
#pragma unroll
        for (int kk = 0; kk < 2; ++kk) {
            v16bf a  = load_frag_a(Qw + kk * 32, QSTR);
            v16bf b0 = load_frag_b(Klds + kk * 32, KSTR);
            v16bf b1 = load_frag_b(Klds + 16 * KSTR + kk * 32, KSTR);
            s0 = wmma_bf16(a, b0, s0);
            s1 = wmma_bf16(a, b1, s1);
        }
#pragma unroll
        for (int r = 0; r < 8; ++r) {
            float a0 = __expf(s0[r] - M[r]) * Zinv[r];   // final attn
            float a1 = __expf(s1[r] - M[r]) * Zinv[r];
            float u0 = __expf(-a0), u1 = __expf(-a1);    // in [e^-1, 1], safe
            U[r] += u0 + u1;
            int prow = r + ((lane < 16) ? 0 : 8);
            int pcol = lane & 15;
            Pw[prow * PSTR + pcol]      = (bf16_t)u0;
            Pw[prow * PSTR + 16 + pcol] = (bf16_t)u1;
        }
        v16bf pa = load_frag_a(Pw, PSTR);
#pragma unroll
        for (int t = 0; t < 4; ++t) {
            v16bf b = load_frag_b(Vlds + t * 16 * VSTR, VSTR);
            Oa[t] = wmma_bf16(pa, b, Oa[t]);
        }
    }

#pragma unroll
    for (int r = 0; r < 8; ++r) {
#pragma unroll
        for (int m = 1; m < 16; m <<= 1) U[r] += __shfl_xor(U[r], m, 32);
    }

    const float s = *sp;
    const int n = nh >> 3, h = nh & 7;
#pragma unroll
    for (int t = 0; t < 4; ++t) {
        int d = h * HD + t * 16 + (lane & 15);
#pragma unroll
        for (int r = 0; r < 8; ++r) {
            int row = rowblk + wave * 16 + r + ((lane < 16) ? 0 : 8);
            float val = s * Oc[t][r] + (1.0f - s) * Oa[t][r] / U[r];
            Out[((size_t)n * TT + row) * DD + d] = val;
        }
    }
}

// ---------------------------------------------------------------------------
extern "C" void kernel_launch(void* const* d_in, const int* in_sizes, int n_in,
                              void* d_out, int out_size, void* d_ws, size_t ws_size,
                              hipStream_t stream) {
    const float* x   = (const float*)d_in[0];
    const float* y   = (const float*)d_in[1];
    const float* iw  = (const float*)d_in[2];
    const float* lw  = (const float*)d_in[3];
    const float* isc = (const float*)d_in[4];
    const float* lsc = (const float*)d_in[5];
    const float* sp  = (const float*)d_in[6];
    float* out = (float*)d_out;

    const size_t S = (size_t)NB * HH * TT * HD;   // per-tensor bf16 elements
    bf16_t* ws = (bf16_t*)d_ws;
    bf16_t* iq = ws;
    bf16_t* ik = iq + S;
    bf16_t* iv = ik + S;
    bf16_t* lq = iv + S;
    bf16_t* lk = lq + S;
    bf16_t* lv = lk + S;
    float* qsi = (float*)(lv + S);
    float* qsl = qsi + HH * HD;

    scale_softmax_kernel<<<dim3(HH, 2), HD, 0, stream>>>(isc, lsc, qsi, qsl);

    qkv_gemm_kernel<<<dim3((NB * TT) / GM, (3 * DD) / GN), 256, 0, stream>>>(
        x, iw, iq, ik, iv, qsi);
    qkv_gemm_kernel<<<dim3((NB * TT) / GM, (3 * DD) / GN), 256, 0, stream>>>(
        y, lw, lq, lk, lv, qsl);

    // x_out = dual_attend(i_q, l_k, Vc=l_v, Vs=i_v)
    attn_kernel<<<dim3(TT / BM, NB * HH), 128, 0, stream>>>(
        iq, lk, lv, iv, out, sp);
    // y_out = dual_attend(l_q, i_k, Vc=i_v, Vs=l_v)
    attn_kernel<<<dim3(TT / BM, NB * HH), 128, 0, stream>>>(
        lq, ik, iv, lv, out + (size_t)NB * TT * DD, sp);
}
